// MutltiHeadAttention_32899449487956
// MI455X (gfx1250) — matrix operand order repaired and checked by reading, not compiled, not run on hardware
//
#include <hip/hip_runtime.h>
#include <math.h>

#define D_MODEL 1024
#define HEADS 16
#define HEAD_DIM 64
#define SEQ 2048
#define BATCH 2

typedef __attribute__((ext_vector_type(16))) __bf16        v16bf;
typedef __attribute__((ext_vector_type(8)))  float         v8f;
typedef __attribute__((ext_vector_type(4)))  __bf16        v4bf;
typedef __attribute__((ext_vector_type(4)))  unsigned int  u32x4;
typedef __attribute__((ext_vector_type(4)))  float         f32x4;
typedef __attribute__((__vector_size__(4 * sizeof(int)))) int i32x4v;

union FragU { v16bf v; u32x4 q[2]; };

#if defined(__AMDGCN__) && __has_builtin(__builtin_amdgcn_global_load_async_to_lds_b128) && \
    __has_builtin(__builtin_amdgcn_s_wait_asynccnt)
#define HAVE_ASYNC_LDS 1
#else
#define HAVE_ASYNC_LDS 0
#endif

// 16-byte global -> LDS copy. Uses the CDNA5 async LDS path when the builtin
// is available (tracked by ASYNCcnt), else a plain b128 LDS store.
__device__ __forceinline__ void cp_async16(__bf16* lds_dst, const __bf16* gsrc) {
#if HAVE_ASYNC_LDS
  __builtin_amdgcn_global_load_async_to_lds_b128(
      (__attribute__((address_space(1))) i32x4v*)gsrc,
      (__attribute__((address_space(3))) i32x4v*)lds_dst, 0, 0);
#else
  *(u32x4*)lds_dst = *(const u32x4*)gsrc;
#endif
}

__device__ __forceinline__ void async_wait_all() {
#if HAVE_ASYNC_LDS
  __builtin_amdgcn_s_wait_asynccnt(0);
#endif
}

// A-matrix fragment (16x32 bf16). Lane l<16 holds row M=l&15, K {k0+0..7, k0+16..23};
// lanes 16-31 hold K {k0+8..15, k0+24..31}. Two contiguous 16B loads per lane.
__device__ __forceinline__ v16bf frag_a(const __bf16* base, int ldm, int row, int k0, int half) {
  FragU u;
  const __bf16* p = base + (size_t)row * ldm + k0 + 8 * half;
  u.q[0] = *(const u32x4*)(p);
  u.q[1] = *(const u32x4*)(p + 16);
  return u.v;
}

// B-matrix fragment (32x16 bf16), base indexed as [n][k]. Lane holds column
// n = col base + lane&15, K {k0+8*half+0..7, k0+16+8*half+0..7} (matches A).
__device__ __forceinline__ v16bf frag_b(const __bf16* base, int ldm, int col, int k0, int half) {
  FragU u;
  const __bf16* p = base + (size_t)col * ldm + k0 + 8 * half;
  u.q[0] = *(const u32x4*)(p);
  u.q[1] = *(const u32x4*)(p + 16);
  return u.v;
}

__device__ __forceinline__ v8f wmma_bf16(v16bf a, v16bf b, v8f c) {
  return __builtin_amdgcn_wmma_f32_16x16x32_bf16(false, a, false, b, (short)0, c, false, false);
}

// ---------------------------------------------------------------------------
// Kernel 0: f32 -> bf16 conversion (one-shot; removes cvt from GEMM hot loop
// and halves repeated global traffic).
// ---------------------------------------------------------------------------
__global__ __launch_bounds__(256) void cvt_kernel(const float* __restrict__ src,
                                                  __bf16* __restrict__ dst, int n) {
  int i = (blockIdx.x * 256 + threadIdx.x) * 4;
  if (i >= n) return;
  f32x4 f = *(const f32x4*)(src + i);
  v4bf o;
  o.x = (__bf16)f.x; o.y = (__bf16)f.y; o.z = (__bf16)f.z; o.w = (__bf16)f.w;
  *(v4bf*)(dst + i) = o;
}

// ---------------------------------------------------------------------------
// Unified GEMM: Y = X(bf16) @ W(bf16)^T + bias, M=4096 N=1024 K=1024.
// 128x128 block tile, 8 waves of 32x64, double-buffered async LDS staging.
// MODE 0: write bf16 scattered to [B,H,L,Dh].  MODE 1: write f32 row-major.
// ---------------------------------------------------------------------------
template <int MODE>
__global__ __launch_bounds__(256) void gemm_kernel(
    const __bf16* __restrict__ X, const __bf16* __restrict__ W,
    const float* __restrict__ bias, void* __restrict__ dstv)
{
  __shared__ __bf16 As[2][128 * 40];   // [m][k], padded stride 40
  __shared__ __bf16 Bs[2][128 * 40];   // [n][k]

  const int mb = blockIdx.y * 128;
  const int nb = blockIdx.x * 128;
  const int t = threadIdx.x;
  const int lane = t & 31;
  const int wave = t >> 5;
  const int wr = wave & 3;     // 4 row slabs of 32
  const int wc = wave >> 2;    // 2 col slabs of 64
  const int half = lane >> 4;
  const int l15 = lane & 15;

  v8f acc[2][4] = {};

  auto stage = [&](int buf, int k0) {
#pragma unroll
    for (int j = 0; j < 2; ++j) {
      int id = t + j * 256;            // 512 16B chunks per matrix
      int row = id >> 2;
      int c = (id & 3) * 8;
      cp_async16(&As[buf][row * 40 + c], X + (size_t)(mb + row) * 1024 + k0 + c);
      cp_async16(&Bs[buf][row * 40 + c], W + (size_t)(nb + row) * 1024 + k0 + c);
    }
  };

  stage(0, 0);
  async_wait_all();
  __syncthreads();

  for (int k0 = 0; k0 < 1024; k0 += 32) {
    const int cur = (k0 >> 5) & 1;
    if (k0 + 32 < 1024) stage(cur ^ 1, k0 + 32);   // prefetch next tile (async)

    v16bf af[2], bfr[4];
#pragma unroll
    for (int mt = 0; mt < 2; ++mt)
      af[mt] = frag_a(As[cur], 40, wr * 32 + mt * 16 + l15, 0, half);
#pragma unroll
    for (int nt = 0; nt < 4; ++nt)
      bfr[nt] = frag_b(Bs[cur], 40, wc * 64 + nt * 16 + l15, 0, half);
#pragma unroll
    for (int mt = 0; mt < 2; ++mt)
#pragma unroll
      for (int nt = 0; nt < 4; ++nt)
        acc[mt][nt] = wmma_bf16(af[mt], bfr[nt], acc[mt][nt]);

    async_wait_all();
    __syncthreads();
  }

#pragma unroll
  for (int mt = 0; mt < 2; ++mt) {
#pragma unroll
    for (int nt = 0; nt < 4; ++nt) {
      int n = nb + wc * 64 + nt * 16 + l15;
      float bb = bias[n];
#pragma unroll
      for (int r = 0; r < 8; ++r) {
        int m = mb + wr * 32 + mt * 16 + half * 8 + r;   // C layout: M = r + 8*half
        if constexpr (MODE == 0) {
          __bf16* dst = (__bf16*)dstv;
          int hh = n >> 6, dh = n & 63;
          int bidx = m >> 11, l = m & (SEQ - 1);
          dst[(((size_t)(bidx * HEADS + hh)) * SEQ + l) * HEAD_DIM + dh] =
              (__bf16)(acc[mt][nt][r] + bb);
        } else {
          float* dst = (float*)dstv;
          dst[(size_t)m * 1024 + n] = acc[mt][nt][r] + bb;
        }
      }
    }
  }
}

// ---------------------------------------------------------------------------
// Flash attention. One block (4 waves) per (b, h, 64-row Q slab). Each wave
// owns a 16-row slab: S = Q K^T, online softmax, O += P V. K staged async.
// ---------------------------------------------------------------------------
__global__ __launch_bounds__(128) void attn_kernel(
    const __bf16* __restrict__ Qh, const __bf16* __restrict__ Kh,
    const __bf16* __restrict__ Vh, const int* __restrict__ mask,
    __bf16* __restrict__ ctx)
{
  __shared__ __bf16 Ks[64 * 72];   // [key][dh]
  __shared__ __bf16 Vt[64 * 72];   // [dh][key]  (transposed for B-fragments)
  __shared__ __bf16 Ps[64 * 72];   // [qrow][key]
  __shared__ int    ms[64];

  const int mblk = blockIdx.x;
  const int h    = blockIdx.y;
  const int b    = blockIdx.z;
  const int t = threadIdx.x;
  const int lane = t & 31;
  const int w = t >> 5;
  const int half = lane >> 4;
  const int l15 = lane & 15;

  const size_t headoff = ((size_t)(b * HEADS + h)) * SEQ * HEAD_DIM;
  const __bf16* qbase = Qh + headoff + (size_t)(mblk * 64 + w * 16) * HEAD_DIM;
  const __bf16* kbase = Kh + headoff;
  const __bf16* vbase = Vh + headoff;

  // Q fragments live in registers for the whole kernel (Dh = 64 -> 2 K-chunks)
  v16bf qa0 = frag_a(qbase, HEAD_DIM, l15, 0, half);
  v16bf qa1 = frag_a(qbase, HEAD_DIM, l15, 32, half);

  float Mr[8], Lsm[8];
  v8f o[4] = {};
#pragma unroll
  for (int r = 0; r < 8; ++r) { Mr[r] = -3.0e38f; Lsm[r] = 0.f; }

  for (int kb = 0; kb < SEQ / 64; ++kb) {
    // K tile: async global->LDS (no VGPR roundtrip); overlaps with V transpose.
#pragma unroll
    for (int j = 0; j < 4; ++j) {
      int id = t + j * 128;        // 512 16B chunks
      int row = id >> 3;           // key 0..63
      int c = (id & 7) * 8;        // dh offset
      cp_async16(&Ks[row * 72 + c], kbase + (size_t)(kb * 64 + row) * HEAD_DIM + c);
    }
    // V tile: load through VGPRs and store transposed [dh][key]
#pragma unroll
    for (int j = 0; j < 4; ++j) {
      int id = t + j * 128;
      int row = id >> 3;
      int c = (id & 7) * 8;
      u32x4 vv = *(const u32x4*)(vbase + (size_t)(kb * 64 + row) * HEAD_DIM + c);
#pragma unroll
      for (int i = 0; i < 8; ++i) {
        unsigned int wd = vv[i >> 1];
        unsigned short hw = (unsigned short)(wd >> (16 * (i & 1)));
        Vt[(c + i) * 72 + row] = __builtin_bit_cast(__bf16, hw);
      }
    }
    if (t < 64) ms[t] = mask[b * SEQ + kb * 64 + t];
    async_wait_all();
    __syncthreads();

    // S = Q @ K^T  (16 x 64 per wave)
    v8f s[4] = {};
#pragma unroll
    for (int j = 0; j < 4; ++j) {
      s[j] = wmma_bf16(qa0, frag_b(Ks, 72, j * 16 + l15, 0, half), s[j]);
      s[j] = wmma_bf16(qa1, frag_b(Ks, 72, j * 16 + l15, 32, half), s[j]);
    }

    // scale + mask + row max (C layout: row = r + 8*half, col = lane&15)
    float mnew[8];
#pragma unroll
    for (int r = 0; r < 8; ++r) mnew[r] = -3.0e38f;
#pragma unroll
    for (int j = 0; j < 4; ++j) {
      float mk = (float)ms[j * 16 + l15] * -1.0e15f;
#pragma unroll
      for (int r = 0; r < 8; ++r) {
        float sv = s[j][r] * 0.125f + mk;
        s[j][r] = sv;
        mnew[r] = fmaxf(mnew[r], sv);
      }
    }
#pragma unroll
    for (int r = 0; r < 8; ++r) {
#pragma unroll
      for (int x = 1; x < 16; x <<= 1)
        mnew[r] = fmaxf(mnew[r], __shfl_xor(mnew[r], x, 32));
      float Mn = fmaxf(Mr[r], mnew[r]);
      float alpha = __expf(Mr[r] - Mn);
      Mr[r] = Mn;
      Lsm[r] *= alpha;
#pragma unroll
      for (int d = 0; d < 4; ++d) o[d][r] *= alpha;
    }

    // P = exp(S - M) -> bf16 LDS (own slab; DS ops in-order per wave)
    float rs[8];
#pragma unroll
    for (int r = 0; r < 8; ++r) rs[r] = 0.f;
#pragma unroll
    for (int j = 0; j < 4; ++j) {
#pragma unroll
      for (int r = 0; r < 8; ++r) {
        float p = __expf(s[j][r] - Mr[r]);
        rs[r] += p;
        Ps[(w * 16 + half * 8 + r) * 72 + j * 16 + l15] = (__bf16)p;
      }
    }
#pragma unroll
    for (int r = 0; r < 8; ++r) {
#pragma unroll
      for (int x = 1; x < 16; x <<= 1) rs[r] += __shfl_xor(rs[r], x, 32);
      Lsm[r] += rs[r];
    }

    // O += P @ V  (K-dim = 64 keys -> two WMMA per output tile)
    const __bf16* pbase = &Ps[w * 16 * 72];
    v16bf pa0 = frag_a(pbase, 72, l15, 0, half);
    v16bf pa1 = frag_a(pbase, 72, l15, 32, half);
#pragma unroll
    for (int d = 0; d < 4; ++d) {
      o[d] = wmma_bf16(pa0, frag_b(Vt, 72, d * 16 + l15, 0, half), o[d]);
      o[d] = wmma_bf16(pa1, frag_b(Vt, 72, d * 16 + l15, 32, half), o[d]);
    }
    __syncthreads();
  }

  // Normalize and write context bf16, merged-head layout [B*L, D_MODEL]
#pragma unroll
  for (int d = 0; d < 4; ++d) {
#pragma unroll
    for (int r = 0; r < 8; ++r) {
      int m = b * SEQ + mblk * 64 + w * 16 + half * 8 + r;
      int col = h * HEAD_DIM + d * 16 + l15;
      ctx[(size_t)m * D_MODEL + col] = (__bf16)(o[d][r] / Lsm[r]);
    }
  }
}

// ---------------------------------------------------------------------------
extern "C" void kernel_launch(void* const* d_in, const int* in_sizes, int n_in,
                              void* d_out, int out_size, void* d_ws, size_t ws_size,
                              hipStream_t stream) {
  const float* q    = (const float*)d_in[0];
  const float* k    = (const float*)d_in[1];
  const float* v    = (const float*)d_in[2];
  const int*   mask = (const int*)d_in[3];
  const float* Wq   = (const float*)d_in[4];
  const float* bq   = (const float*)d_in[5];
  const float* Wk   = (const float*)d_in[6];
  const float* bk   = (const float*)d_in[7];
  const float* Wv   = (const float*)d_in[8];
  const float* bv   = (const float*)d_in[9];
  const float* Wo   = (const float*)d_in[10];
  const float* bo   = (const float*)d_in[11];

  const size_t XE = (size_t)BATCH * SEQ * D_MODEL;   // 4M elems
  const size_t WE = (size_t)D_MODEL * D_MODEL;       // 1M elems

  __bf16* base = (__bf16*)d_ws;
  __bf16* xq = base;
  __bf16* xk = xq + XE;
  __bf16* xv = xk + XE;
  __bf16* wq = xv + XE;
  __bf16* wk = wq + WE;
  __bf16* wv = wk + WE;
  __bf16* wo = wv + WE;
  __bf16* qh = wo + WE;
  __bf16* kh = qh + XE;
  __bf16* vh = kh + XE;
  __bf16* ctx = vh + XE;                              // total 32M elems = 64 MiB

  const int gX = (int)(XE / 4 / 256);   // 4096 blocks
  const int gW = (int)(WE / 4 / 256);   // 1024 blocks
  cvt_kernel<<<gX, 256, 0, stream>>>(q, xq, (int)XE);
  cvt_kernel<<<gX, 256, 0, stream>>>(k, xk, (int)XE);
  cvt_kernel<<<gX, 256, 0, stream>>>(v, xv, (int)XE);
  cvt_kernel<<<gW, 256, 0, stream>>>(Wq, wq, (int)WE);
  cvt_kernel<<<gW, 256, 0, stream>>>(Wk, wk, (int)WE);
  cvt_kernel<<<gW, 256, 0, stream>>>(Wv, wv, (int)WE);
  cvt_kernel<<<gW, 256, 0, stream>>>(Wo, wo, (int)WE);

  gemm_kernel<0><<<dim3(8, 32), 256, 0, stream>>>(xq, wq, bq, qh);
  gemm_kernel<0><<<dim3(8, 32), 256, 0, stream>>>(xk, wk, bk, kh);
  gemm_kernel<0><<<dim3(8, 32), 256, 0, stream>>>(xv, wv, bv, vh);

  attn_kernel<<<dim3(SEQ / 64, HEADS, BATCH), 128, 0, stream>>>(qh, kh, vh, mask, ctx);

  gemm_kernel<1><<<dim3(8, 32), 256, 0, stream>>>(ctx, wo, bo, d_out);
}
